// ProTACT_60696477827631
// MI455X (gfx1250) — matrix operand order repaired and checked by reading, not compiled, not run on hardware
//
#include <hip/hip_runtime.h>
#include <hip/hip_bf16.h>

// ---------------- model dims ----------------
#define B_SZ   128
#define MAXNUM 64
#define MAXLEN 50
#define EMBD   50
#define FILT   100
#define DOUT   9
#define LINGD  51
#define READD  35
#define KW     5
#define LPOS   46      // MAXLEN - KW + 1
#define E_DIM  186     // FILT + LING + READ
#define E_PAD  192
#define FL     400     // 4 * LSTMU

typedef __attribute__((ext_vector_type(16))) _Float16 v16h;
typedef __attribute__((ext_vector_type(8)))  float    v8f;

__device__ __forceinline__ float sigm(float x) { return 1.0f / (1.0f + expf(-x)); }

// ---------------- WMMA helpers (f32 = f16 x f16, 16x16x32) ----------------
// A-matrix 16x32 per-lane layout (ISA 7.12.2):
//   lanes 0-15 : M=lane,    halves map K = {0..7, 16..23}
//   lanes16-31 : M=lane-16, halves map K = {8..15, 24..31}
// (two contiguous 8-half runs -> 2x b128 loads)
__device__ __forceinline__ v16h wmma_load_a(const _Float16* A, int ld, int row0, int k0) {
  const int l = threadIdx.x & 31;
  const _Float16* p = A + (row0 + (l & 15)) * ld + k0 + ((l >> 4) << 3);
  v16h a;
#pragma unroll
  for (int h = 0; h < 16; ++h) {
    const int k = (h & 7) + ((h >> 3) << 4);
    a[h] = p[k];
  }
  return a;
}

// B-matrix 32x16 where memory holds B^T row-major [N][K]:
//   lane reads two contiguous 8-half runs along K -> 2x b128 loads
//   lanes 0-15 hold K = k0+{0..7,16..23} ; lanes 16-31 hold K = k0+{8..15,24..31} ; N = lane&15
__device__ __forceinline__ v16h wmma_load_bt(const _Float16* BT, int ld, int k0, int col0) {
  const int l = threadIdx.x & 31;
  const _Float16* p = BT + (col0 + (l & 15)) * ld + k0 + ((l >> 4) << 3);
  v16h b;
#pragma unroll
  for (int h = 0; h < 16; ++h) b[h] = p[(h & 7) + ((h >> 3) << 4)];
  return b;
}

__device__ __forceinline__ v8f wmma16x16x32(v16h a, v16h b, v8f c) {
  return __builtin_amdgcn_wmma_f32_16x16x32_f16(false, a, false, b, (short)0, c, false, false);
}

// C/D element owned by (lane, i): M = i + 8*(lane>=16), N = lane&15
__device__ __forceinline__ void wmma_cd(int i, int& m, int& n) {
  const int l = threadIdx.x & 31;
  m = i + ((l >> 4) << 3);
  n = l & 15;
}

// ================= K0: pad LSTM weights to f16 [d][400][128] (natural [N][K] layout) =================
__global__ __launch_bounds__(256) void k_prep_weights(
    const float* __restrict__ tWih, const float* __restrict__ tWhh,
    const float* __restrict__ cWih,
    _Float16* __restrict__ wihP, _Float16* __restrict__ whhP, _Float16* __restrict__ cwihP) {
  int i = blockIdx.x * 256 + threadIdx.x;
  const int total = DOUT * FL * 128;
  if (i >= total) return;
  int d = i / (FL * 128);
  int r = i % (FL * 128);
  int n = r / 128, k = r % 128;
  float a = 0.f, h = 0.f, c = 0.f;
  if (k < FILT) {
    a = tWih[(d * FL + n) * FILT + k];
    h = tWhh[(d * FL + n) * FILT + k];
    c = cWih[(d * FL + n) * FILT + k];
  }
  wihP[i] = (_Float16)a; whhP[i] = (_Float16)h; cwihP[i] = (_Float16)c;
}

// ================= K1: embed + conv1d (implicit GEMM) + att_pool =================
// one block per (b, n) sentence; writes pos_avg[b][n][100]
__global__ __launch_bounds__(256) void k_conv_pool(
    const int* __restrict__ pos_input, const float* __restrict__ pos_emb,
    const float* __restrict__ convW, const float* __restrict__ convb,
    const float* __restrict__ eattW, const float* __restrict__ eattv,
    float* __restrict__ posavg) {
  const int bn = blockIdx.x;            // b*64 + n
  const int b = bn >> 6, n = bn & 63;
  const int tid = threadIdx.x, wave = tid >> 5;

  __shared__ __align__(16) _Float16 shW[112 * 256];    // conv A, reused as att W^T
  __shared__ __align__(16) _Float16 shImT[48 * 256];   // im2col TRANSPOSED [t][K]
  __shared__ __align__(16) _Float16 shZ[48 * 128];     // z[t][f] (K-padded)
  __shared__ __align__(16) _Float16 shX[50 * 50];      // embedded [e][t]
  __shared__ int   shIdx[64];
  __shared__ float shV[112];
  __shared__ float shLogit[64];

  if (tid < 50) shIdx[tid] = pos_input[b * (MAXNUM * MAXLEN) + n * MAXLEN + tid];
  __syncthreads();
  for (int i = tid; i < 50 * 50; i += 256) {
    int e = i / 50, t = i % 50;
    int idx = shIdx[t];
    shX[i] = (_Float16)((idx != 0) ? pos_emb[idx * EMBD + e] : 0.0f);
  }
  for (int i = tid; i < 112 * 256; i += 256) {
    int f = i >> 8, k = i & 255;
    shW[i] = (_Float16)((f < FILT && k < 250) ? convW[f * 250 + k] : 0.0f);
  }
  for (int i = tid; i < 48 * 128; i += 256) shZ[i] = (_Float16)0.0f;
  __syncthreads();
  for (int i = tid; i < 48 * 256; i += 256) {
    int t = i >> 8, k = i & 255;
    _Float16 v = (_Float16)0.0f;
    if (k < 250 && t < LPOS) v = shX[(k / 5) * 50 + (t + k % 5)];
    shImT[i] = v;
  }
  __syncthreads();
  // GEMM1: Z[f][t] = W[100x250] * Im[250x46], store as shZ[t][f]
  for (int tile = wave; tile < 21; tile += 8) {
    int mt = tile / 3, nt = tile % 3;
    v8f acc = {};
#pragma unroll
    for (int kc = 0; kc < 8; ++kc) {
      v16h a  = wmma_load_a(shW, 256, mt * 16, kc * 32);
      v16h bm = wmma_load_bt(shImT, 256, kc * 32, nt * 16);
      acc = wmma16x16x32(a, bm, acc);
    }
#pragma unroll
    for (int i = 0; i < 8; ++i) {
      int m, nn; wmma_cd(i, m, nn);
      int f = mt * 16 + m, t = nt * 16 + nn;
      float v = (f < FILT) ? (acc[i] + convb[f]) : 0.0f;
      shZ[t * 128 + f] = (_Float16)v;
    }
  }
  __syncthreads();
  // att weights (TRANSPOSED: shW[g][k]) for this sentence position n
  for (int i = tid; i < 112 * 128; i += 256) {
    int g = i >> 7, k = i & 127;
    shW[i] = (_Float16)((k < FILT && g < FILT) ? eattW[n * FILT * FILT + k * FILT + g] : 0.0f);
  }
  if (tid < 112) shV[tid] = (tid < FILT) ? eattv[n * FILT + tid] : 0.0f;
  if (tid < 64) shLogit[tid] = 0.0f;
  __syncthreads();
  // GEMM2: s = tanh(z @ W); logits via LDS float atomics
  for (int tile = wave; tile < 21; tile += 8) {
    int mt = tile / 7, nt = tile % 7;
    v8f acc = {};
#pragma unroll
    for (int kc = 0; kc < 4; ++kc) {
      v16h a  = wmma_load_a(shZ, 128, mt * 16, kc * 32);
      v16h bm = wmma_load_bt(shW, 128, kc * 32, nt * 16);
      acc = wmma16x16x32(a, bm, acc);
    }
#pragma unroll
    for (int i = 0; i < 8; ++i) {
      int m, nn; wmma_cd(i, m, nn);
      int t = mt * 16 + m, g = nt * 16 + nn;
      if (t < LPOS && g < FILT) atomicAdd(&shLogit[t], tanhf(acc[i]) * shV[g]);
    }
  }
  __syncthreads();
  if (tid == 0) {
    float mx = -1e30f;
    for (int t = 0; t < LPOS; ++t) mx = fmaxf(mx, shLogit[t]);
    float s = 0.0f;
    for (int t = 0; t < LPOS; ++t) { float e = expf(shLogit[t] - mx); shLogit[t] = e; s += e; }
    float inv = 1.0f / s;
    for (int t = 0; t < LPOS; ++t) shLogit[t] *= inv;
  }
  __syncthreads();
  if (tid < FILT) {
    float s = 0.0f;
    for (int t = 0; t < LPOS; ++t) s += shLogit[t] * (float)shZ[t * 128 + tid];
    posavg[bn * FILT + tid] = s;
  }
}

// ================= K2: per-trait 2-head self attention over 64 positions =================
// grid (128, 9); writes mha_y[d][b][t][100] (f16)
__global__ __launch_bounds__(256) void k_trait_mha(
    const float* __restrict__ posavg,
    const float* __restrict__ Wq, const float* __restrict__ bq,
    const float* __restrict__ Wk, const float* __restrict__ bk,
    const float* __restrict__ Wv, const float* __restrict__ bv,
    const float* __restrict__ Wo, const float* __restrict__ bo,
    _Float16* __restrict__ mha_y) {
  const int b = blockIdx.x, d = blockIdx.y;
  const int tid = threadIdx.x, wave = tid >> 5;

  __shared__ __align__(16) _Float16 shX[64 * 128];    // also aliased as f32 scores
  __shared__ __align__(16) _Float16 shW[112 * 128];   // W^T [g][k]
  __shared__ __align__(16) _Float16 shQ[64 * 128];
  __shared__ __align__(16) _Float16 shK[64 * 128];
  __shared__ __align__(16) _Float16 shVT[128 * 64];   // V TRANSPOSED [g][t]
  __shared__ __align__(16) _Float16 shO[64 * 128];
  __shared__ __align__(16) _Float16 shSh[64 * 64];
  float* shS = (float*)shX;                            // reuse after projections

  for (int i = tid; i < 64 * 128; i += 256) {
    int t = i >> 7, k = i & 127;
    shX[i]  = (_Float16)((k < FILT) ? posavg[(b * 64 + t) * FILT + k] : 0.0f);
    shQ[i]  = (_Float16)0.0f;
    shK[i]  = (_Float16)0.0f;
    shVT[i] = (_Float16)0.0f;
    shO[i]  = (_Float16)0.0f;
  }
  const float* Wsrc[3] = { Wq + d * FILT * FILT, Wk + d * FILT * FILT, Wv + d * FILT * FILT };
  const float* bsrc[3] = { bq + d * FILT, bk + d * FILT, bv + d * FILT };
  for (int p = 0; p < 3; ++p) {
    _Float16* dstp = (p == 0) ? shQ : shK;    // runtime select; p==2 writes shVT
    __syncthreads();
    for (int i = tid; i < 112 * 128; i += 256) {
      int g = i >> 7, k = i & 127;
      shW[i] = (_Float16)((k < FILT && g < FILT) ? Wsrc[p][k * FILT + g] : 0.0f);
    }
    __syncthreads();
    for (int tile = wave; tile < 28; tile += 8) {
      int mt = tile / 7, nt = tile % 7;
      v8f acc = {};
#pragma unroll
      for (int kc = 0; kc < 4; ++kc) {
        v16h a  = wmma_load_a(shX, 128, mt * 16, kc * 32);
        v16h bm = wmma_load_bt(shW, 128, kc * 32, nt * 16);
        acc = wmma16x16x32(a, bm, acc);
      }
#pragma unroll
      for (int i = 0; i < 8; ++i) {
        int m, nn; wmma_cd(i, m, nn);
        int t = mt * 16 + m, g = nt * 16 + nn;
        _Float16 v = (_Float16)((g < FILT) ? (acc[i] + bsrc[p][g]) : 0.0f);
        if (p == 2) shVT[g * 64 + t] = v;
        else        dstp[t * 128 + g] = v;
      }
    }
  }
  __syncthreads();
  const float scale = 0.14142135623f;   // 1/sqrt(50)
  for (int h = 0; h < 2; ++h) {
    // S = Qh @ Kh^T  (K-dim 64 from col h*50: h=1 pads with zero cols 100..113, h=0 takes head-1 cols 50..63)
    for (int tile = wave; tile < 16; tile += 8) {
      int mt = tile / 4, nt = tile % 4;
      v8f acc = {};
#pragma unroll
      for (int kc = 0; kc < 2; ++kc) {
        v16h a  = wmma_load_a(shQ + h * 50, 128, mt * 16, kc * 32);
        v16h bm = wmma_load_bt(shK + h * 50, 128, kc * 32, nt * 16);
        acc = wmma16x16x32(a, bm, acc);
      }
#pragma unroll
      for (int i = 0; i < 8; ++i) {
        int m, nn; wmma_cd(i, m, nn);
        shS[(mt * 16 + m) * 64 + nt * 16 + nn] = acc[i] * scale;
      }
    }
    __syncthreads();
    if (tid < 64) {
      float mx = -1e30f;
      for (int j = 0; j < 64; ++j) mx = fmaxf(mx, shS[tid * 64 + j]);
      float s = 0.0f;
      for (int j = 0; j < 64; ++j) { float e = expf(shS[tid * 64 + j] - mx); shS[tid * 64 + j] = e; s += e; }
      float inv = 1.0f / s;
      for (int j = 0; j < 64; ++j) shSh[tid * 64 + j] = (_Float16)(shS[tid * 64 + j] * inv);
    }
    __syncthreads();
    // O_h = softmax(S) @ V_h  (B from V^T rows: contiguous)
    for (int tile = wave; tile < 16; tile += 8) {
      int mt = tile / 4, nt = tile % 4;
      v8f acc = {};
#pragma unroll
      for (int kc = 0; kc < 2; ++kc) {
        v16h a  = wmma_load_a(shSh, 64, mt * 16, kc * 32);
        v16h bm = wmma_load_bt(shVT + h * 50 * 64, 64, kc * 32, nt * 16);
        acc = wmma16x16x32(a, bm, acc);
      }
#pragma unroll
      for (int i = 0; i < 8; ++i) {
        int m, nn; wmma_cd(i, m, nn);
        int t = mt * 16 + m, g = nt * 16 + nn;
        if (g < 50) shO[t * 128 + h * 50 + g] = (_Float16)acc[i];
      }
    }
    __syncthreads();
  }
  // Y = O @ Wo + bo
  for (int i = tid; i < 112 * 128; i += 256) {
    int g = i >> 7, k = i & 127;
    shW[i] = (_Float16)((k < FILT && g < FILT) ? Wo[d * FILT * FILT + k * FILT + g] : 0.0f);
  }
  __syncthreads();
  for (int tile = wave; tile < 28; tile += 8) {
    int mt = tile / 7, nt = tile % 7;
    v8f acc = {};
#pragma unroll
    for (int kc = 0; kc < 4; ++kc) {
      v16h a  = wmma_load_a(shO, 128, mt * 16, kc * 32);
      v16h bm = wmma_load_bt(shW, 128, kc * 32, nt * 16);
      acc = wmma16x16x32(a, bm, acc);
    }
#pragma unroll
    for (int i = 0; i < 8; ++i) {
      int m, nn; wmma_cd(i, m, nn);
      int t = mt * 16 + m, g = nt * 16 + nn;
      if (g < FILT)
        mha_y[(((size_t)d * B_SZ + b) * 64 + t) * FILT + g] =
            (_Float16)(acc[i] + bo[d * FILT + g]);
    }
  }
}

// ================= K3: per-trait LSTM, batched gate GEMMs [64x400 = 64x128 * 128x400] =================
// grid (9, 2): trait d, batch half. hseq[d][b][t][100] (f16)
__global__ __launch_bounds__(256) void k_trait_lstm(
    const _Float16* __restrict__ mha_y, const _Float16* __restrict__ wihP,
    const _Float16* __restrict__ whhP, const float* __restrict__ bih,
    const float* __restrict__ bhh, _Float16* __restrict__ hseq) {
  const int d = blockIdx.x, bbase = blockIdx.y * 64;
  const int tid = threadIdx.x, wave = tid >> 5;
  __shared__ __align__(16) _Float16 shXt[64 * 128];
  __shared__ __align__(16) _Float16 shH[64 * 128];
  __shared__ __align__(16) _Float16 shG[64 * FL];
  __shared__ float shC[64 * FILT];
  __shared__ float shB[FL];

  for (int i = tid; i < 64 * 128; i += 256) shH[i] = (_Float16)0.0f;
  for (int i = tid; i < 64 * FILT; i += 256) shC[i] = 0.0f;
  for (int i = tid; i < FL; i += 256) shB[i] = bih[d * FL + i] + bhh[d * FL + i];
  const _Float16* Wi = wihP + (size_t)d * FL * 128;   // [gate n][k] row-major
  const _Float16* Wh = whhP + (size_t)d * FL * 128;

  for (int t = 0; t < 64; ++t) {
    for (int i = tid; i < 64 * 128; i += 256) {
      int bb = i >> 7, k = i & 127;
      shXt[i] = (k < FILT)
          ? mha_y[(((size_t)d * B_SZ + bbase + bb) * 64 + t) * FILT + k]
          : (_Float16)0.0f;
    }
    __syncthreads();
    for (int tile = wave; tile < 100; tile += 8) {     // 4 mtiles x 25 ntiles
      int mt = tile / 25, nt = tile % 25;
      v8f acc = {};
#pragma unroll
      for (int kc = 0; kc < 4; ++kc) {
        v16h a  = wmma_load_a(shXt, 128, mt * 16, kc * 32);
        v16h bm = wmma_load_bt(Wi, 128, kc * 32, nt * 16);
        acc = wmma16x16x32(a, bm, acc);
      }
#pragma unroll
      for (int kc = 0; kc < 4; ++kc) {
        v16h a  = wmma_load_a(shH, 128, mt * 16, kc * 32);
        v16h bm = wmma_load_bt(Wh, 128, kc * 32, nt * 16);
        acc = wmma16x16x32(a, bm, acc);
      }
#pragma unroll
      for (int i = 0; i < 8; ++i) {
        int m, nn; wmma_cd(i, m, nn);
        int bb = mt * 16 + m, g = nt * 16 + nn;
        shG[bb * FL + g] = (_Float16)(acc[i] + shB[g]);
      }
    }
    __syncthreads();
    for (int i = tid; i < 64 * FILT; i += 256) {
      int bb = i / FILT, j = i % FILT;
      float ig = sigm((float)shG[bb * FL + j]);
      float fg = sigm((float)shG[bb * FL + 100 + j]);
      float gg = tanhf((float)shG[bb * FL + 200 + j]);
      float og = sigm((float)shG[bb * FL + 300 + j]);
      float c = fg * shC[i] + ig * gg;
      shC[i] = c;
      float h = og * tanhf(c);
      shH[bb * 128 + j] = (_Float16)h;
      hseq[(((size_t)d * B_SZ + bbase + bb) * 64 + t) * FILT + j] = (_Float16)h;
    }
    __syncthreads();
  }
}

// ================= K4: per-trait attention pooling over LSTM outputs =================
// grid (128, 9); reps[d][b][100]
__global__ __launch_bounds__(256) void k_trait_pool(
    const _Float16* __restrict__ hseq, const float* __restrict__ aW,
    const float* __restrict__ av, float* __restrict__ reps) {
  const int b = blockIdx.x, d = blockIdx.y;
  const int tid = threadIdx.x, wave = tid >> 5;
  __shared__ __align__(16) _Float16 shHx[64 * 128];
  __shared__ __align__(16) _Float16 shW[112 * 128];   // aW^T [g][k]
  __shared__ float shV[112];
  __shared__ float shLogit[64];

  for (int i = tid; i < 64 * 128; i += 256) {
    int t = i >> 7, k = i & 127;
    shHx[i] = (k < FILT) ? hseq[(((size_t)d * B_SZ + b) * 64 + t) * FILT + k] : (_Float16)0.0f;
  }
  for (int i = tid; i < 112 * 128; i += 256) {
    int g = i >> 7, k = i & 127;
    shW[i] = (_Float16)((k < FILT && g < FILT) ? aW[d * FILT * FILT + k * FILT + g] : 0.0f);
  }
  if (tid < 112) shV[tid] = (tid < FILT) ? av[d * FILT + tid] : 0.0f;
  if (tid < 64) shLogit[tid] = 0.0f;
  __syncthreads();
  for (int tile = wave; tile < 28; tile += 8) {
    int mt = tile / 7, nt = tile % 7;
    v8f acc = {};
#pragma unroll
    for (int kc = 0; kc < 4; ++kc) {
      v16h a  = wmma_load_a(shHx, 128, mt * 16, kc * 32);
      v16h bm = wmma_load_bt(shW, 128, kc * 32, nt * 16);
      acc = wmma16x16x32(a, bm, acc);
    }
#pragma unroll
    for (int i = 0; i < 8; ++i) {
      int m, nn; wmma_cd(i, m, nn);
      int t = mt * 16 + m, g = nt * 16 + nn;
      if (g < FILT) atomicAdd(&shLogit[t], tanhf(acc[i]) * shV[g]);
    }
  }
  __syncthreads();
  if (tid == 0) {
    float mx = -1e30f;
    for (int t = 0; t < 64; ++t) mx = fmaxf(mx, shLogit[t]);
    float s = 0.0f;
    for (int t = 0; t < 64; ++t) { float e = expf(shLogit[t] - mx); shLogit[t] = e; s += e; }
    float inv = 1.0f / s;
    for (int t = 0; t < 64; ++t) shLogit[t] *= inv;
  }
  __syncthreads();
  if (tid < FILT) {
    float s = 0.0f;
    for (int t = 0; t < 64; ++t) s += shLogit[t] * (float)shHx[t * 128 + tid];
    reps[((size_t)d * B_SZ + b) * FILT + tid] = s;
  }
}

// ================= K5: cross projection + single LSTM step (h0=c0=0) =================
// grid (9, 2); crossv[d][b][100]
__global__ __launch_bounds__(256) void k_cross(
    const float* __restrict__ reps, const float* __restrict__ Wv, const float* __restrict__ bv,
    const float* __restrict__ Wo, const float* __restrict__ bo,
    const _Float16* __restrict__ cwihP, const float* __restrict__ bih,
    const float* __restrict__ bhh, float* __restrict__ crossv) {
  const int d = blockIdx.x, bbase = blockIdx.y * 64;
  const int tid = threadIdx.x, wave = tid >> 5;
  __shared__ __align__(16) _Float16 shA[64 * 128];     // reps, later o2
  __shared__ __align__(16) _Float16 shBuf[64 * 128];   // o1
  __shared__ __align__(16) _Float16 shW[112 * 128];    // W^T [g][k]
  __shared__ __align__(16) _Float16 shG[64 * FL];
  __shared__ float shB[FL];

  for (int i = tid; i < 64 * 128; i += 256) {
    int bb = i >> 7, k = i & 127;
    shA[i]   = (k < FILT) ? (_Float16)reps[((size_t)d * B_SZ + bbase + bb) * FILT + k] : (_Float16)0.0f;
    shBuf[i] = (_Float16)0.0f;
  }
  for (int i = tid; i < FL; i += 256) shB[i] = bih[d * FL + i] + bhh[d * FL + i];
  for (int i = tid; i < 112 * 128; i += 256) {
    int g = i >> 7, k = i & 127;
    shW[i] = (_Float16)((k < FILT && g < FILT) ? Wv[d * FILT * FILT + k * FILT + g] : 0.0f);
  }
  __syncthreads();
  // o1 = reps @ Wv + bv
  for (int tile = wave; tile < 28; tile += 8) {
    int mt = tile / 7, nt = tile % 7;
    v8f acc = {};
#pragma unroll
    for (int kc = 0; kc < 4; ++kc)
      acc = wmma16x16x32(wmma_load_a(shA, 128, mt * 16, kc * 32),
                         wmma_load_bt(shW, 128, kc * 32, nt * 16), acc);
#pragma unroll
    for (int i = 0; i < 8; ++i) {
      int m, nn; wmma_cd(i, m, nn);
      int bb = mt * 16 + m, g = nt * 16 + nn;
      shBuf[bb * 128 + g] = (_Float16)((g < FILT) ? (acc[i] + bv[d * FILT + g]) : 0.0f);
    }
  }
  __syncthreads();
  for (int i = tid; i < 112 * 128; i += 256) {
    int g = i >> 7, k = i & 127;
    shW[i] = (_Float16)((k < FILT && g < FILT) ? Wo[d * FILT * FILT + k * FILT + g] : 0.0f);
  }
  __syncthreads();
  // o2 = o1 @ Wo + bo   (into shA)
  for (int tile = wave; tile < 28; tile += 8) {
    int mt = tile / 7, nt = tile % 7;
    v8f acc = {};
#pragma unroll
    for (int kc = 0; kc < 4; ++kc)
      acc = wmma16x16x32(wmma_load_a(shBuf, 128, mt * 16, kc * 32),
                         wmma_load_bt(shW, 128, kc * 32, nt * 16), acc);
#pragma unroll
    for (int i = 0; i < 8; ++i) {
      int m, nn; wmma_cd(i, m, nn);
      int bb = mt * 16 + m, g = nt * 16 + nn;
      shA[bb * 128 + g] = (_Float16)((g < FILT) ? (acc[i] + bo[d * FILT + g]) : 0.0f);
    }
  }
  __syncthreads();
  // G = o2 @ WihT + (bih + bhh)
  const _Float16* Wi = cwihP + (size_t)d * FL * 128;
  for (int tile = wave; tile < 100; tile += 8) {
    int mt = tile / 25, nt = tile % 25;
    v8f acc = {};
#pragma unroll
    for (int kc = 0; kc < 4; ++kc)
      acc = wmma16x16x32(wmma_load_a(shA, 128, mt * 16, kc * 32),
                         wmma_load_bt(Wi, 128, kc * 32, nt * 16), acc);
#pragma unroll
    for (int i = 0; i < 8; ++i) {
      int m, nn; wmma_cd(i, m, nn);
      int bb = mt * 16 + m, g = nt * 16 + nn;
      shG[bb * FL + g] = (_Float16)(acc[i] + shB[g]);
    }
  }
  __syncthreads();
  for (int i = tid; i < 64 * FILT; i += 256) {
    int bb = i / FILT, j = i % FILT;
    float ig = sigm((float)shG[bb * FL + j]);
    float gg = tanhf((float)shG[bb * FL + 200 + j]);
    float og = sigm((float)shG[bb * FL + 300 + j]);
    float c = ig * gg;                     // f-gate * c0 = 0
    float h = og * tanhf(c);
    crossv[((size_t)d * B_SZ + bbase + bb) * FILT + j] = h;
  }
}

// ================= K6a: assemble rep[b][d][192] = [cross | ling | read | 0pad] =================
__global__ __launch_bounds__(256) void k_build_rep(
    const float* __restrict__ crossv, const float* __restrict__ ling,
    const float* __restrict__ readf, float* __restrict__ rep32,
    _Float16* __restrict__ rep16) {
  int i = blockIdx.x * 256 + threadIdx.x;
  const int total = B_SZ * DOUT * E_PAD;
  if (i >= total) return;
  int b = i / (DOUT * E_PAD);
  int r = i % (DOUT * E_PAD);
  int d = r / E_PAD, e = r % E_PAD;
  float v = 0.0f;
  if (e < FILT) v = crossv[((size_t)d * B_SZ + b) * FILT + e];
  else if (e < FILT + LINGD) v = ling[b * LINGD + (e - FILT)];
  else if (e < E_DIM) v = readf[b * READD + (e - FILT - LINGD)];
  rep32[i] = v;
  rep16[i] = (_Float16)v;
}

// ================= K6b: fatt projections Q/K/V: [1152x192] = [1152x192]x[192x192] =================
// grid (9, 3): trait d, which-of-qkv
__global__ __launch_bounds__(256) void k_fatt_proj(
    const _Float16* __restrict__ rep16,
    const float* __restrict__ Wq, const float* __restrict__ bq,
    const float* __restrict__ Wk, const float* __restrict__ bk,
    const float* __restrict__ Wv, const float* __restrict__ bv,
    float* __restrict__ qp, float* __restrict__ kp, float* __restrict__ vp) {
  const int d = blockIdx.x, p = blockIdx.y;
  const int tid = threadIdx.x, wave = tid >> 5;
  const float* W    = (p == 0 ? Wq : (p == 1 ? Wk : Wv)) + (size_t)d * E_DIM * E_DIM;
  const float* bias = (p == 0 ? bq : (p == 1 ? bk : bv)) + (size_t)d * E_DIM;
  float* out        = (p == 0 ? qp : (p == 1 ? kp : vp)) + (size_t)d * (B_SZ * DOUT) * E_PAD;
  __shared__ __align__(16) _Float16 shW[E_PAD * E_PAD];   // W^T [g][k]
  for (int i = tid; i < E_PAD * E_PAD; i += 256) {
    int g = i / E_PAD, k = i % E_PAD;
    shW[i] = (_Float16)((k < E_DIM && g < E_DIM) ? W[k * E_DIM + g] : 0.0f);
  }
  __syncthreads();
  for (int tile = wave; tile < 72 * 12; tile += 8) {   // 1152/16 x 192/16
    int mt = tile / 12, nt = tile % 12;
    v8f acc = {};
#pragma unroll
    for (int kc = 0; kc < 6; ++kc) {
      v16h a  = wmma_load_a(rep16, E_PAD, mt * 16, kc * 32);   // A straight from global
      v16h bm = wmma_load_bt(shW, E_PAD, kc * 32, nt * 16);
      acc = wmma16x16x32(a, bm, acc);
    }
#pragma unroll
    for (int i = 0; i < 8; ++i) {
      int m, nn; wmma_cd(i, m, nn);
      int row = mt * 16 + m, g = nt * 16 + nn;
      out[(size_t)row * E_PAD + g] = (g < E_DIM) ? (acc[i] + bias[g]) : 0.0f;
    }
  }
}

// ================= K7: neighbor softmax, context, Wo proj, dense sigmoid =================
// grid 128 (b); out[b][d]
__global__ __launch_bounds__(256) void k_fatt_final(
    const float* __restrict__ qp, const float* __restrict__ kp, const float* __restrict__ vp,
    const float* __restrict__ Wo, const float* __restrict__ bo,
    const float* __restrict__ rep32, const float* __restrict__ denseW,
    const float* __restrict__ denseb, float* __restrict__ out) {
  const int b = blockIdx.x, tid = threadIdx.x;
  __shared__ float shLog[DOUT][8];
  __shared__ float shCtx[DOUT][E_PAD];
  __shared__ float shAtt[DOUT][E_PAD];
  const float rs = rsqrtf((float)E_DIM);

  for (int i = tid; i < DOUT * 8; i += 256) {
    int d = i / 8, j = i % 8;
    int dp = (j < d) ? j : j + 1;
    const float* q  = qp + ((size_t)d * (B_SZ * DOUT) + b * DOUT + d)  * E_PAD;
    const float* kk = kp + ((size_t)d * (B_SZ * DOUT) + b * DOUT + dp) * E_PAD;
    float s = 0.0f;
    for (int e = 0; e < E_DIM; ++e) s += q[e] * kk[e];
    shLog[d][j] = s * rs;
  }
  __syncthreads();
  if (tid < DOUT) {
    float mx = -1e30f;
    for (int j = 0; j < 8; ++j) mx = fmaxf(mx, shLog[tid][j]);
    float s = 0.0f;
    for (int j = 0; j < 8; ++j) { float e = expf(shLog[tid][j] - mx); shLog[tid][j] = e; s += e; }
    float inv = 1.0f / s;
    for (int j = 0; j < 8; ++j) shLog[tid][j] *= inv;
  }
  __syncthreads();
  for (int i = tid; i < DOUT * E_PAD; i += 256) {
    int d = i / E_PAD, e = i % E_PAD;
    float s = 0.0f;
    for (int j = 0; j < 8; ++j) {
      int dp = (j < d) ? j : j + 1;
      s += shLog[d][j] * vp[((size_t)d * (B_SZ * DOUT) + b * DOUT + dp) * E_PAD + e];
    }
    shCtx[d][e] = s;
  }
  __syncthreads();
  for (int i = tid; i < DOUT * E_DIM; i += 256) {
    int d = i / E_DIM, e = i % E_DIM;
    const float* W = Wo + (size_t)d * E_DIM * E_DIM;
    float s = bo[d * E_DIM + e];
    for (int k = 0; k < E_DIM; ++k) s += shCtx[d][k] * W[k * E_DIM + e];
    shAtt[d][e] = s;
  }
  __syncthreads();
  if (tid < DOUT) {
    int d = tid;
    float s = denseb[d];
    const float* Wd = denseW + d * (2 * E_DIM);
    const float* rp = rep32 + ((size_t)b * DOUT + d) * E_PAD;
    for (int f = 0; f < E_DIM; ++f) s += rp[f] * Wd[f];
    for (int f = 0; f < E_DIM; ++f) s += shAtt[d][f] * Wd[E_DIM + f];
    out[b * DOUT + d] = 1.0f / (1.0f + expf(-s));
  }
}

// ================= host launcher =================
extern "C" void kernel_launch(void* const* d_in, const int* in_sizes, int n_in,
                              void* d_out, int out_size, void* d_ws, size_t ws_size,
                              hipStream_t stream) {
  (void)in_sizes; (void)n_in; (void)out_size; (void)ws_size;
  const int*   pos_input = (const int*)d_in[0];
  const float* ling      = (const float*)d_in[3];
  const float* readf     = (const float*)d_in[4];
  const float* pos_emb   = (const float*)d_in[5];
  const float* econvW    = (const float*)d_in[8];
  const float* econvb    = (const float*)d_in[9];
  const float* eattW     = (const float*)d_in[12];
  const float* eattv     = (const float*)d_in[13];
  const float* tWq  = (const float*)d_in[16];
  const float* tbq  = (const float*)d_in[17];
  const float* tWk  = (const float*)d_in[18];
  const float* tbk  = (const float*)d_in[19];
  const float* tWv  = (const float*)d_in[20];
  const float* tbv  = (const float*)d_in[21];
  const float* tWo  = (const float*)d_in[22];
  const float* tbo  = (const float*)d_in[23];
  const float* tWih = (const float*)d_in[24];
  const float* tWhh = (const float*)d_in[25];
  const float* tbih = (const float*)d_in[26];
  const float* tbhh = (const float*)d_in[27];
  const float* taW  = (const float*)d_in[28];
  const float* tav  = (const float*)d_in[29];
  const float* cWv  = (const float*)d_in[44];
  const float* cbv  = (const float*)d_in[45];
  const float* cWo  = (const float*)d_in[46];
  const float* cbo  = (const float*)d_in[47];
  const float* cWih = (const float*)d_in[48];
  const float* cbih = (const float*)d_in[50];
  const float* cbhh = (const float*)d_in[51];
  const float* fWq  = (const float*)d_in[54];
  const float* fbq  = (const float*)d_in[55];
  const float* fWk  = (const float*)d_in[56];
  const float* fbk  = (const float*)d_in[57];
  const float* fWv  = (const float*)d_in[58];
  const float* fbv  = (const float*)d_in[59];
  const float* fWo  = (const float*)d_in[60];
  const float* fbo  = (const float*)d_in[61];
  const float* denseW = (const float*)d_in[62];
  const float* denseb = (const float*)d_in[63];
  float* out = (float*)d_out;

  // workspace layout (256B aligned)
  constexpr size_t ALN = 256;
  auto up = [](size_t x) { return (x + ALN - 1) & ~(ALN - 1); };
  size_t o = 0;
  const size_t o_posavg = o; o += up((size_t)B_SZ * MAXNUM * FILT * 4);
  const size_t o_mhay   = o; o += up((size_t)DOUT * B_SZ * MAXNUM * FILT * 2);
  const size_t o_hseq   = o; o += up((size_t)DOUT * B_SZ * MAXNUM * FILT * 2);
  const size_t o_reps   = o; o += up((size_t)DOUT * B_SZ * FILT * 4);
  const size_t o_cross  = o; o += up((size_t)DOUT * B_SZ * FILT * 4);
  const size_t o_wihP   = o; o += up((size_t)DOUT * FL * 128 * 2);
  const size_t o_whhP   = o; o += up((size_t)DOUT * FL * 128 * 2);
  const size_t o_cwihP  = o; o += up((size_t)DOUT * FL * 128 * 2);
  const size_t o_rep16  = o; o += up((size_t)B_SZ * DOUT * E_PAD * 2);
  const size_t o_rep32  = o; o += up((size_t)B_SZ * DOUT * E_PAD * 4);
  const size_t o_qp     = o; o += up((size_t)DOUT * B_SZ * DOUT * E_PAD * 4);
  const size_t o_kp     = o; o += up((size_t)DOUT * B_SZ * DOUT * E_PAD * 4);
  const size_t o_vp     = o; o += up((size_t)DOUT * B_SZ * DOUT * E_PAD * 4);

  char* w = (char*)d_ws;
  float*    posavg = (float*)(w + o_posavg);
  _Float16* mhay   = (_Float16*)(w + o_mhay);
  _Float16* hseq   = (_Float16*)(w + o_hseq);
  float*    reps   = (float*)(w + o_reps);
  float*    crossv = (float*)(w + o_cross);
  _Float16* wihP   = (_Float16*)(w + o_wihP);
  _Float16* whhP   = (_Float16*)(w + o_whhP);
  _Float16* cwihP  = (_Float16*)(w + o_cwihP);
  _Float16* rep16  = (_Float16*)(w + o_rep16);
  float*    rep32  = (float*)(w + o_rep32);
  float*    qp     = (float*)(w + o_qp);
  float*    kp     = (float*)(w + o_kp);
  float*    vp     = (float*)(w + o_vp);

  k_prep_weights<<<(DOUT * FL * 128 + 255) / 256, 256, 0, stream>>>(
      tWih, tWhh, cWih, wihP, whhP, cwihP);
  k_conv_pool<<<B_SZ * MAXNUM, 256, 0, stream>>>(
      pos_input, pos_emb, econvW, econvb, eattW, eattv, posavg);
  k_trait_mha<<<dim3(B_SZ, DOUT), 256, 0, stream>>>(
      posavg, tWq, tbq, tWk, tbk, tWv, tbv, tWo, tbo, mhay);
  k_trait_lstm<<<dim3(DOUT, 2), 256, 0, stream>>>(
      mhay, wihP, whhP, tbih, tbhh, hseq);
  k_trait_pool<<<dim3(B_SZ, DOUT), 256, 0, stream>>>(hseq, taW, tav, reps);
  k_cross<<<dim3(DOUT, 2), 256, 0, stream>>>(
      reps, cWv, cbv, cWo, cbo, cwihP, cbih, cbhh, crossv);
  k_build_rep<<<(B_SZ * DOUT * E_PAD + 255) / 256, 256, 0, stream>>>(
      crossv, ling, readf, rep32, rep16);
  k_fatt_proj<<<dim3(DOUT, 3), 256, 0, stream>>>(
      rep16, fWq, fbq, fWk, fbk, fWv, fbv, qp, kp, vp);
  k_fatt_final<<<B_SZ, 256, 0, stream>>>(
      qp, kp, vp, fWo, fbo, rep32, denseW, denseb, out);
}
